// Channel_Attention_83502754169314
// MI455X (gfx1250) — matrix operand order repaired and checked by reading, not compiled, not run on hardware
//
#include <hip/hip_runtime.h>
#include <hip/hip_bf16.h>
#include <stdint.h>

typedef __attribute__((ext_vector_type(16))) __bf16 v16bf;
typedef __attribute__((ext_vector_type(8)))  float  v8f;

#define CH    180
#define CH3   540
#define HDS   6
#define DH    30          // channels per head
#define PSZ   7           // window size
#define PP    49          // window pixels
#define IMG   448
#define HWSZ  200704      // 448*448
#define NBAT  2
#define NWIN  64          // windows per side

#define KPADD 192         // K padded to 6*32
#define MP_QKV 640        // 540 padded to 5*128
#define MP_PROJ 256       // 180 padded to 2*128

// ---------- bf16 helpers: native hardware converts (RNE) ----------
static __device__ __forceinline__ __bf16 f2bf(float f) { return (__bf16)f; }
static __device__ __forceinline__ unsigned short f2bfbits(float f) {
  union { __bf16 b; unsigned short s; } c; c.b = (__bf16)f;
  return c.s;
}
static __device__ __forceinline__ float bf2f(__bf16 x) { return (float)x; }
static __device__ __forceinline__ v8f zero8() {
  v8f z;
  #pragma unroll
  for (int i = 0; i < 8; ++i) z[i] = 0.f;
  return z;
}

// ---------------------------------------------------------------------------
// Kernel 1: pack fp32 weights -> zero-padded bf16 [Mpad][192] in workspace
// ---------------------------------------------------------------------------
__global__ void pack_weights_kernel(const float* __restrict__ wq, const float* __restrict__ wp,
                                    __bf16* __restrict__ wq_p, __bf16* __restrict__ wp_p) {
  int i = blockIdx.x * 256 + threadIdx.x;
  if (i < MP_QKV * KPADD) {
    int m = i / KPADD, k = i % KPADD;
    float f = (m < CH3 && k < CH) ? wq[m * CH + k] : 0.f;
    wq_p[i] = f2bf(f);
  }
  if (i < MP_PROJ * KPADD) {
    int m = i / KPADD, k = i % KPADD;
    float f = (m < CH && k < CH) ? wp[m * CH + k] : 0.f;
    wp_p[i] = f2bf(f);
  }
}

// ---------------------------------------------------------------------------
// Kernel 2/4: WMMA GEMM  Out[M, B*HW] = W[M,180] * X[180, B*HW]
//   256 threads = 8 waves; block tile 128(M) x 128(N); K chunks of 32 (180->192)
//   A tile staged with async global->LDS DMA (bf16 copy, 16B/lane)
//   B tile staged branch-free (clamped index + select), transposed Bs[n][k]
// ---------------------------------------------------------------------------
#define TM 128
#define TN 128
#define KC 32
#define KPAD 8
#define NKC 6

template<bool B_IS_BF16, bool OUT_BF16, bool OUT_GUARD>
__global__ __launch_bounds__(256)
void gemm_wmma_kernel(const __bf16* __restrict__ Wb,   // padded [Mpad][KPADD]
                      const void* __restrict__ Bsrc,
                      void* __restrict__ Out, int M, int MOutStride)
{
  __shared__ __bf16 As[TM][KC + KPAD];     // A tile: M x K   (row stride 80B, 16B-aligned)
  __shared__ __bf16 Bs[TN][KC + KPAD];     // B tile transposed: N x K

  const int tid  = threadIdx.x;
  const int lane = tid & 31;
  const int wave = tid >> 5;
  const int wm   = wave >> 1;              // 0..3  (M)
  const int wn   = wave & 1;               // 0..1  (N)
  const int hf   = lane >> 4;
  const int r    = lane & 15;

  const int       m0      = blockIdx.y * TM;
  const long long n0      = (long long)blockIdx.x * TN;  // HW % 128 == 0
  const int       bb      = (int)(n0 / HWSZ);
  const int       hw_base = (int)(n0 % HWSZ);

  const float*  Bf  = (const float*)Bsrc;
  const __bf16* Bbf = (const __bf16*)Bsrc;

  v8f acc[2][4];
  #pragma unroll
  for (int mi = 0; mi < 2; ++mi)
    #pragma unroll
    for (int ni = 0; ni < 4; ++ni) acc[mi][ni] = zero8();

  // lane->K maps (contiguous): A and B: VGPR v holds K=2v,2v+1 (lanes 0-15) / K=16+2v,16+2v+1 (lanes 16-31).
  int kA[8], kB[8];
  #pragma unroll
  for (int v = 0; v < 8; ++v) {
    kA[v] = 2 * v + 16 * hf;
    kB[v] = 2 * v + 16 * hf;
  }

  #pragma unroll
  for (int kc = 0; kc < NKC; ++kc) {
    // ---- A tile: async DMA global->LDS, 16 bytes per lane, no guards (padded) ----
    #pragma unroll
    for (int it = 0; it < 2; ++it) {
      int task = it * 256 + tid;                  // 512 transfers of 16B = 8KB tile
      int i  = task & 127;
      int kk = (task >> 7) * 8;
      const __bf16* gp = Wb + (size_t)(m0 + i) * KPADD + (kc * KC + kk);
      unsigned lds_off = (unsigned)(uintptr_t)(&As[i][kk]);
      unsigned long long ga = (unsigned long long)(uintptr_t)gp;
      asm volatile("global_load_async_to_lds_b128 %0, %1, off"
                   :: "v"(lds_off), "v"(ga) : "memory");
    }

    // ---- B tile: branch-free clamped loads, 4 elems/thread-task, b64 LDS store ----
    #pragma unroll
    for (int it = 0; it < 4; ++it) {
      int task = it * 256 + tid;                  // 1024 tasks x 4 elems
      int j  = task & 127;
      int kk = (task >> 7) * 4;
      int gk0 = kc * KC + kk;
      unsigned short h[4];
      #pragma unroll
      for (int q = 0; q < 4; ++q) {
        int gk  = gk0 + q;
        int gkc = (gk < CH) ? gk : (CH - 1);      // clamp -> unconditional load
        size_t off = ((size_t)(bb * CH + gkc)) * HWSZ + (size_t)hw_base + j;
        if (B_IS_BF16) {
          unsigned short s = *(const unsigned short*)&Bbf[off];
          h[q] = (gk < CH) ? s : (unsigned short)0;
        } else {
          float f = Bf[off];
          f = (gk < CH) ? f : 0.f;
          h[q] = f2bfbits(f);
        }
      }
      unsigned long long packed = (unsigned long long)h[0]
                                | ((unsigned long long)h[1] << 16)
                                | ((unsigned long long)h[2] << 32)
                                | ((unsigned long long)h[3] << 48);
      *(unsigned long long*)(&Bs[j][kk]) = packed;   // kk%4==0 -> 8B aligned
    }

    // prefetch next K-chunk activation rows (global_prefetch_b8)
    if (kc + 1 < NKC) {
      size_t off = ((size_t)(bb * CH + (kc + 1) * KC)) * HWSZ + (size_t)hw_base + (tid & 127);
      if (B_IS_BF16) __builtin_prefetch((const void*)&Bbf[off], 0, 1);
      else           __builtin_prefetch((const void*)&Bf[off],  0, 1);
    }

    asm volatile("s_wait_asynccnt 0x0" ::: "memory");   // drain A-tile DMA
    __syncthreads();

    // ---- fragments from LDS (contiguous K runs -> ds_load_b128) ----
    v16bf a[2], bfr[4];
    #pragma unroll
    for (int mi = 0; mi < 2; ++mi) {
      int row = wm * 32 + mi * 16 + r;
      #pragma unroll
      for (int v = 0; v < 8; ++v) {
        a[mi][2 * v]     = As[row][kA[v]];
        a[mi][2 * v + 1] = As[row][kA[v] + 1];
      }
    }
    #pragma unroll
    for (int ni = 0; ni < 4; ++ni) {
      int col = wn * 64 + ni * 16 + r;
      #pragma unroll
      for (int v = 0; v < 8; ++v) {
        bfr[ni][2 * v]     = Bs[col][kB[v]];
        bfr[ni][2 * v + 1] = Bs[col][kB[v] + 1];
      }
    }

    // ---- 8 WMMAs per wave per K-chunk ----
    #pragma unroll
    for (int mi = 0; mi < 2; ++mi)
      #pragma unroll
      for (int ni = 0; ni < 4; ++ni)
        acc[mi][ni] = __builtin_amdgcn_wmma_f32_16x16x32_bf16(
            false, a[mi], false, bfr[ni], (short)0, acc[mi][ni], false, false);

    __syncthreads();
  }

  // ---- store D ----
  #pragma unroll
  for (int mi = 0; mi < 2; ++mi) {
    #pragma unroll
    for (int ni = 0; ni < 4; ++ni) {
      int colL = wn * 64 + ni * 16 + r;
      #pragma unroll
      for (int i = 0; i < 8; ++i) {
        int gm = m0 + wm * 32 + mi * 16 + hf * 8 + i;
        if (!OUT_GUARD || gm < M) {
          size_t off = ((size_t)(bb * MOutStride + gm)) * HWSZ + (size_t)hw_base + colL;
          if (OUT_BF16) ((__bf16*)Out)[off] = f2bf(acc[mi][ni][i]);
          else          ((float*) Out)[off] = acc[mi][ni][i];
        }
      }
    }
  }
}

// ---------------------------------------------------------------------------
// Kernel 3: fused dwconv3(SAME) + window rearrange + l2norm + channel attention
//   one block per (batch, window_y, window_x, head); qkv stride padded to 640 ch
// ---------------------------------------------------------------------------
#define PPAD 52
__global__ __launch_bounds__(256)
void attn_kernel(const __bf16* __restrict__ qkv, const float* __restrict__ wdw,
                 const float* __restrict__ temp, __bf16* __restrict__ out)
{
  int id   = blockIdx.x;
  int head = id % HDS;  id /= HDS;
  int wx   = id % NWIN; id /= NWIN;
  int wy   = id % NWIN; id /= NWIN;
  int b    = id;

  __shared__ float qs[DH][PPAD];
  __shared__ float ks[DH][PPAD];
  __shared__ float vs[DH][PPAD];
  __shared__ float at[DH][32];

  const int tid = threadIdx.x;

  // ---- depthwise 3x3 conv, branch-free (clamped coords + weight select) ----
  for (int idx = tid; idx < 3 * DH * PP; idx += 256) {
    int p = idx % PP;
    int c = (idx / PP) % DH;
    int t = idx / (PP * DH);
    int ch = t * CH + head * DH + c;
    int py = p / PSZ, px = p % PSZ;
    int gy = wy * PSZ + py, gx = wx * PSZ + px;
    const float*  w9   = wdw + ch * 9;
    const __bf16* base = qkv + ((size_t)(b * MP_QKV + ch)) * HWSZ;
    float s = 0.f;
    #pragma unroll
    for (int ky = 0; ky < 3; ++ky) {
      #pragma unroll
      for (int kx = 0; kx < 3; ++kx) {
        int yy = gy + ky - 1, xx = gx + kx - 1;
        int yc = min(max(yy, 0), IMG - 1);
        int xc = min(max(xx, 0), IMG - 1);
        float w = (yy == yc && xx == xc) ? w9[ky * 3 + kx] : 0.f;   // zero-pad
        s += w * bf2f(base[yc * IMG + xc]);
      }
    }
    if      (t == 0) qs[c][p] = s;
    else if (t == 1) ks[c][p] = s;
    else             vs[c][p] = s;
  }
  __syncthreads();

  // ---- L2 normalize rows of q and k ----
  if (tid < 2 * DH) {
    float (*m)[PPAD] = (tid < DH) ? qs : ks;
    int c = (tid < DH) ? tid : tid - DH;
    float ss = 0.f;
    for (int p = 0; p < PP; ++p) ss += m[c][p] * m[c][p];
    float inv = 1.f / fmaxf(sqrtf(ss), 1e-12f);
    for (int p = 0; p < PP; ++p) m[c][p] *= inv;
  }
  __syncthreads();

  // ---- attn logits ----
  float tscale = temp[head];
  for (int idx = tid; idx < DH * DH; idx += 256) {
    int e = idx % DH, d = idx / DH;
    float s = 0.f;
    for (int p = 0; p < PP; ++p) s += qs[d][p] * ks[e][p];
    at[d][e] = s * tscale;
  }
  __syncthreads();

  // ---- row softmax ----
  if (tid < DH) {
    float mx = -3.4e38f;
    for (int e = 0; e < DH; ++e) mx = fmaxf(mx, at[tid][e]);
    float sum = 0.f;
    for (int e = 0; e < DH; ++e) { float ex = __expf(at[tid][e] - mx); at[tid][e] = ex; sum += ex; }
    float inv = 1.f / sum;
    for (int e = 0; e < DH; ++e) at[tid][e] *= inv;
  }
  __syncthreads();

  // ---- out = attn @ v -> bf16 ----
  for (int idx = tid; idx < DH * PP; idx += 256) {
    int p = idx % PP, d = idx / PP;
    float s = 0.f;
    for (int e = 0; e < DH; ++e) s += at[d][e] * vs[e][p];
    int ch = head * DH + d;
    int py = p / PSZ, px = p % PSZ;
    out[((size_t)(b * CH + ch)) * HWSZ + (size_t)(wy * PSZ + py) * IMG + (wx * PSZ + px)] = f2bf(s);
  }
}

// ---------------------------------------------------------------------------
// launcher
// ---------------------------------------------------------------------------
extern "C" void kernel_launch(void* const* d_in, const int* in_sizes, int n_in,
                              void* d_out, int out_size, void* d_ws, size_t ws_size,
                              hipStream_t stream) {
  const float* x           = (const float*)d_in[0];
  const float* temperature = (const float*)d_in[1];
  const float* w_qkv       = (const float*)d_in[2];
  const float* w_dw        = (const float*)d_in[3];
  const float* w_proj      = (const float*)d_in[4];
  float*       out         = (float*)d_out;
  char*        ws          = (char*)d_ws;

  const size_t sz_qkv = (size_t)NBAT * MP_QKV * HWSZ * sizeof(__bf16);  // padded qkv, ~490 MB
  const size_t sz_att = (size_t)NBAT * CH     * HWSZ * sizeof(__bf16);  // ~138 MB
  const size_t sz_wq  = (size_t)MP_QKV * KPADD * sizeof(__bf16);

  __bf16* qkv_pre  = (__bf16*)(ws);
  __bf16* attn_out = (__bf16*)(ws + sz_qkv);
  __bf16* wq_p     = (__bf16*)(ws + sz_qkv + sz_att);
  __bf16* wp_p     = (__bf16*)(ws + sz_qkv + sz_att + sz_wq);

  // 1. zero-padded weight pack
  pack_weights_kernel<<<(MP_QKV * KPADD + 255) / 256, 256, 0, stream>>>(w_qkv, w_proj, wq_p, wp_p);

  // 2. qkv = Wqkv(540x180) * X(180, B*HW) -> bf16 (640-ch padded, unguarded stores)
  {
    dim3 grid((NBAT * HWSZ) / TN, MP_QKV / TM);   // 3136 x 5
    gemm_wmma_kernel<false, true, false><<<grid, 256, 0, stream>>>(
        wq_p, (const void*)x, (void*)qkv_pre, MP_QKV, MP_QKV);
  }

  // 3. fused dwconv + window channel-attention -> bf16
  attn_kernel<<<NBAT * NWIN * NWIN * HDS, 256, 0, stream>>>(qkv_pre, w_dw, temperature, attn_out);

  // 4. out = Wproj(180x180) * attn(180, B*HW) -> fp32 d_out (guarded rows)
  {
    dim3 grid((NBAT * HWSZ) / TN, MP_PROJ / TM);  // 3136 x 2
    gemm_wmma_kernel<true, false, true><<<grid, 256, 0, stream>>>(
        wp_p, (const void*)attn_out, (void*)out, CH, CH);
  }
}
